// Krylov_37366215475923
// MI455X (gfx1250) — matrix operand order repaired and checked by reading, not compiled, not run on hardware
//
#include <hip/hip_runtime.h>
#include <hip/hip_bf16.h>
#include <math.h>

// Problem constants (match reference)
#define BG 16      // graphs
#define NN 2048    // nodes
#define FF 64      // input feature dim
#define HH 32      // hidden
#define ODIM 16    // output dim
#define DCHUNK 16  // row chunks for degree reduction (NN/128)

typedef __attribute__((ext_vector_type(16))) __bf16 v16bf;
typedef __attribute__((ext_vector_type(8)))  __bf16 v8bf;
typedef __attribute__((ext_vector_type(4)))  __bf16 v4bf;
typedef __attribute__((ext_vector_type(8)))  float  v8f;

// ---------------------------------------------------------------------------
// Degree reduction pass 1: each block owns (b, 128-row chunk), accumulates
// column partial sums in registers (coalesced row-major reads), writes
// dpart[b][chunk][n].  grid = BG*DCHUNK, block = 256.
// ---------------------------------------------------------------------------
__global__ __launch_bounds__(256) void deg_partial_kernel(const float* __restrict__ adj,
                                                          float* __restrict__ dpart) {
  int b  = blockIdx.x / DCHUNK;
  int ic = blockIdx.x % DCHUNK;
  const float* base = adj + (size_t)b * NN * NN + (size_t)ic * 128 * NN;
  float s[8];
  #pragma unroll
  for (int j = 0; j < 8; ++j) s[j] = 0.f;
  for (int i = 0; i < 128; ++i) {
    const float* row = base + (size_t)i * NN;
    #pragma unroll
    for (int j = 0; j < 8; ++j) s[j] += row[threadIdx.x + j * 256];
  }
  float* op = dpart + ((size_t)b * DCHUNK + ic) * NN;
  #pragma unroll
  for (int j = 0; j < 8; ++j) op[threadIdx.x + j * 256] = s[j];
}

// Degree pass 2: d[b,n] = rsqrt( sum_c dpart[b][c][n] ).  grid = BG*NN/256.
__global__ void deg_finish_kernel(const float* __restrict__ dpart,
                                  float* __restrict__ d) {
  int idx = blockIdx.x * blockDim.x + threadIdx.x;  // b*N + n
  int b = idx / NN, n = idx % NN;
  const float* p = dpart + (size_t)b * DCHUNK * NN + n;
  float s = 0.f;
  #pragma unroll
  for (int c = 0; c < DCHUNK; ++c) s += p[(size_t)c * NN];
  d[idx] = rsqrtf(s);
}

// ---------------------------------------------------------------------------
// A_bf16[b,i,j] = bf16( adj[b,i,j] * d[b,i] * d[b,j] )   (float4 granularity)
// ---------------------------------------------------------------------------
__global__ void norm_kernel(const float* __restrict__ adj,
                            const float* __restrict__ d,
                            __bf16* __restrict__ A) {
  size_t idx = (size_t)blockIdx.x * blockDim.x + threadIdx.x;  // one float4
  size_t j4   = idx % (NN / 4);
  size_t rest = idx / (NN / 4);
  int i = (int)(rest % NN);
  int b = (int)(rest / NN);
  float4 a  = reinterpret_cast<const float4*>(adj)[idx];
  float4 dj = reinterpret_cast<const float4*>(d + (size_t)b * NN)[j4];
  float  di = d[(size_t)b * NN + i];
  v4bf o;
  o[0] = (__bf16)(a.x * di * dj.x);
  o[1] = (__bf16)(a.y * di * dj.y);
  o[2] = (__bf16)(a.z * di * dj.z);
  o[3] = (__bf16)(a.w * di * dj.w);
  reinterpret_cast<v4bf*>(A)[idx] = o;
}

// ---------------------------------------------------------------------------
// Transpose+convert: in f32 [B, N, f]  ->  out bf16 [B, f, N]
// 32x32 tiles through padded LDS; block = (32, 8)
// ---------------------------------------------------------------------------
__global__ void cvtT_kernel(const float* __restrict__ in,
                            __bf16* __restrict__ out, int f) {
  __shared__ float tile[32][33];
  int ftiles = f >> 5;
  int bidx = blockIdx.x;
  int ft   = bidx % ftiles;
  int rest = bidx / ftiles;
  int nt   = rest % (NN / 32);
  int b    = rest / (NN / 32);
  int tx = threadIdx.x, ty = threadIdx.y;
  int n0 = nt * 32, f0 = ft * 32;
  const float* ip = in + (size_t)b * NN * f;
  #pragma unroll
  for (int r = 0; r < 4; ++r) {
    int row = ty + r * 8;
    tile[row][tx] = ip[(size_t)(n0 + row) * f + f0 + tx];
  }
  __syncthreads();
  __bf16* op = out + (size_t)b * f * NN;
  #pragma unroll
  for (int r = 0; r < 4; ++r) {
    int c = ty + r * 8;
    op[(size_t)(f0 + c) * NN + n0 + tx] = (__bf16)tile[tx][c];
  }
}

// ---------------------------------------------------------------------------
// Y[b] = A[b] @ X[b].  A bf16 [N,N] row-major, XT bf16 [f,N] (X transposed),
// Y f32 [N,f] row-major.  Each wave: 32 rows (two 16-row A frags) x F cols.
// Ping-pong double-buffered K loop, 2x unrolled: each phase loads one
// fragment set while the WMMAs consume the other -> no register copies,
// no full loadcnt drains.  grid = B*(N/128), block = 128 (4 waves).
// ---------------------------------------------------------------------------
__device__ __forceinline__ v16bf comb16(v8bf lo, v8bf hi) {
  v16bf r;
  #pragma unroll
  for (int i = 0; i < 8; ++i) { r[i] = lo[i]; r[i + 8] = hi[i]; }
  return r;
}

template <int NT>
__global__ __launch_bounds__(128) void gemm_kernel(const __bf16* __restrict__ A,
                                                   const __bf16* __restrict__ XT,
                                                   float* __restrict__ Y) {
  constexpr int F = NT * 16;
  int wave = threadIdx.x >> 5;            // 0..3
  int lane = threadIdx.x & 31;
  int b  = blockIdx.x >> 4;               // 16 row-blocks of 128 rows per graph
  int rb = blockIdx.x & 15;
  int rowBase = rb * 128 + wave * 32;     // 32 rows per wave
  int l15 = lane & 15;
  int hi  = lane >> 4;

  // A fragment layout (16-bit A 16x32): lane<16 -> row=lane, K {0..7,16..23};
  // lane>=16 -> row=lane-16, K {8..15,24..31}.  Fold +hi*8 into base.
  const __bf16* arow0 = A + (size_t)b * NN * NN + (size_t)(rowBase + l15) * NN + hi * 8;
  const __bf16* arow1 = arow0 + (size_t)16 * NN;
  // B fragment (32x16): lane<16 -> col=lane K {0..7,16..23}; lane>=16 -> K {8..15,24..31}.
  const __bf16* bbase = XT + (size_t)b * F * NN + (size_t)l15 * NN + hi * 8;

  v8f zero = 0.f;
  v8f acc0[NT], acc1[NT];
  #pragma unroll
  for (int t = 0; t < NT; ++t) { acc0[t] = zero; acc1[t] = zero; }

  // Two fragment sets (ping-pong).
  v8bf pA0l, pA0h, pA1l, pA1h, pBl[NT], pBh[NT];   // set P
  v8bf qA0l, qA0h, qA1l, qA1h, qBl[NT], qBh[NT];   // set Q

  auto load_p = [&](int k) {
    pA0l = *reinterpret_cast<const v8bf*>(arow0 + k);
    pA0h = *reinterpret_cast<const v8bf*>(arow0 + k + 16);
    pA1l = *reinterpret_cast<const v8bf*>(arow1 + k);
    pA1h = *reinterpret_cast<const v8bf*>(arow1 + k + 16);
    #pragma unroll
    for (int t = 0; t < NT; ++t) {
      pBl[t] = *reinterpret_cast<const v8bf*>(bbase + (size_t)t * 16 * NN + k);
      pBh[t] = *reinterpret_cast<const v8bf*>(bbase + (size_t)t * 16 * NN + k + 16);
    }
  };
  auto load_q = [&](int k) {
    qA0l = *reinterpret_cast<const v8bf*>(arow0 + k);
    qA0h = *reinterpret_cast<const v8bf*>(arow0 + k + 16);
    qA1l = *reinterpret_cast<const v8bf*>(arow1 + k);
    qA1h = *reinterpret_cast<const v8bf*>(arow1 + k + 16);
    #pragma unroll
    for (int t = 0; t < NT; ++t) {
      qBl[t] = *reinterpret_cast<const v8bf*>(bbase + (size_t)t * 16 * NN + k);
      qBh[t] = *reinterpret_cast<const v8bf*>(bbase + (size_t)t * 16 * NN + k + 16);
    }
  };
  auto compute_p = [&]() {
    v16bf af0 = comb16(pA0l, pA0h);
    v16bf af1 = comb16(pA1l, pA1h);
    #pragma unroll
    for (int t = 0; t < NT; ++t) {
      v16bf bf = comb16(pBl[t], pBh[t]);
      acc0[t] = __builtin_amdgcn_wmma_f32_16x16x32_bf16(
          false, af0, false, bf, (short)0, acc0[t], false, false);
      acc1[t] = __builtin_amdgcn_wmma_f32_16x16x32_bf16(
          false, af1, false, bf, (short)0, acc1[t], false, false);
    }
  };
  auto compute_q = [&]() {
    v16bf af0 = comb16(qA0l, qA0h);
    v16bf af1 = comb16(qA1l, qA1h);
    #pragma unroll
    for (int t = 0; t < NT; ++t) {
      v16bf bf = comb16(qBl[t], qBh[t]);
      acc0[t] = __builtin_amdgcn_wmma_f32_16x16x32_bf16(
          false, af0, false, bf, (short)0, acc0[t], false, false);
      acc1[t] = __builtin_amdgcn_wmma_f32_16x16x32_bf16(
          false, af1, false, bf, (short)0, acc1[t], false, false);
    }
  };

  // Prologue: K-tile 0 into set P.
  load_p(0);

  // Main loop: 2 K-tiles (64) per iteration, stop before the last pair.
  int k0 = 0;
  #pragma unroll 1
  for (; k0 < NN - 64; k0 += 64) {
    load_q(k0 + 32);                               // in flight during compute_p
    __builtin_prefetch(arow0 + k0 + 96, 0, 1);     // A stream ahead
    __builtin_prefetch(arow1 + k0 + 96, 0, 1);
    compute_p();
    load_p(k0 + 64);                               // in flight during compute_q
    __builtin_prefetch(arow0 + k0 + 128, 0, 1);
    __builtin_prefetch(arow1 + k0 + 128, 0, 1);
    compute_q();
  }
  // Tail: k0 == NN-64.  Set P holds tile NN-64; load NN-32 into Q.
  load_q(NN - 32);
  compute_p();
  compute_q();

  // C/D layout: VGPR v, lane -> row m = v + hi*8, col n = lane&15
  float* yrow0 = Y + ((size_t)b * NN + rowBase) * F;
  float* yrow1 = yrow0 + (size_t)16 * F;
  #pragma unroll
  for (int t = 0; t < NT; ++t) {
    #pragma unroll
    for (int v = 0; v < 8; ++v) {
      int m = v + hi * 8;
      yrow0[(size_t)m * F + t * 16 + l15] = acc0[t][v];
      yrow1[(size_t)m * F + t * 16 + l15] = acc1[t][v];
    }
  }
}

// ---------------------------------------------------------------------------
// out[node][o] = act( bias[o] + sum_s sum_k src_s[node][k] * W[s*f+k][o] )
// W cached in LDS; block = (H, 256/H); grid = B*N / blockDim.y
// act: 0 = tanh, 1 = relu
// ---------------------------------------------------------------------------
__global__ void proj_kernel(const float* __restrict__ s0, const float* __restrict__ s1,
                            const float* __restrict__ s2, const float* __restrict__ s3,
                            const float* __restrict__ W,  const float* __restrict__ bias,
                            float* __restrict__ out, int f, int nsrc, int H, int relu_act) {
  __shared__ float Wl[8192];  // max 4*64*32 floats = 32 KB
  int tot = nsrc * f * H;
  int tid = threadIdx.y * blockDim.x + threadIdx.x;
  int nth = blockDim.x * blockDim.y;
  for (int i = tid; i < tot; i += nth) Wl[i] = W[i];
  __syncthreads();

  int o = threadIdx.x;
  size_t node = (size_t)blockIdx.x * blockDim.y + threadIdx.y;
  const float* srcs[4] = {s0, s1, s2, s3};
  float acc = bias ? bias[o] : 0.f;
  for (int s = 0; s < nsrc; ++s) {
    const float* row = srcs[s] + node * (size_t)f;
    const float* w   = Wl + s * f * H + o;
    #pragma unroll 8
    for (int k = 0; k < f; ++k) acc += row[k] * w[k * H];
  }
  out[node * (size_t)H + o] = relu_act ? fmaxf(acc, 0.f) : tanhf(acc);
}

// ---------------------------------------------------------------------------
extern "C" void kernel_launch(void* const* d_in, const int* in_sizes, int n_in,
                              void* d_out, int out_size, void* d_ws, size_t ws_size,
                              hipStream_t stream) {
  (void)in_sizes; (void)n_in; (void)out_size; (void)ws_size;
  const float* feats = (const float*)d_in[0];   // [B,N,64]
  const float* adj   = (const float*)d_in[1];   // [B,N,N]
  const float* W0    = (const float*)d_in[2];   // [256,32]
  const float* W1    = (const float*)d_in[3];   // [128,32]
  const float* W2    = (const float*)d_in[4];   // [128,32]
  const float* Wm    = (const float*)d_in[5];   // [96,16]
  const float* bm    = (const float*)d_in[6];   // [16]
  float* out = (float*)d_out;                   // [B,N,16]

  char* ws = (char*)d_ws;
  size_t off = 0;
  auto alloc = [&](size_t bytes) -> void* {
    void* p = ws + off;
    off = (off + bytes + 255) & ~(size_t)255;
    return p;
  };
  float*  dpart = (float*)alloc((size_t)BG * DCHUNK * NN * sizeof(float));  // 2 MB
  float*  ddeg  = (float*)alloc((size_t)BG * NN * sizeof(float));
  __bf16* Abf   = (__bf16*)alloc((size_t)BG * NN * NN * sizeof(__bf16));    // 128 MB
  __bf16* XT    = (__bf16*)alloc((size_t)BG * FF * NN * sizeof(__bf16));    // 4 MB
  float*  Y1    = (float*)alloc((size_t)BG * NN * FF * sizeof(float));      // 8 MB
  float*  Y2    = (float*)alloc((size_t)BG * NN * FF * sizeof(float));
  float*  Y3    = (float*)alloc((size_t)BG * NN * FF * sizeof(float));
  float*  h0    = (float*)alloc((size_t)BG * NN * HH * sizeof(float));
  float*  h1    = (float*)alloc((size_t)BG * NN * HH * sizeof(float));
  float*  h2    = (float*)alloc((size_t)BG * NN * HH * sizeof(float));

  // Degree (two-pass deterministic) + normalized bf16 adjacency
  deg_partial_kernel<<<BG * DCHUNK, 256, 0, stream>>>(adj, dpart);
  deg_finish_kernel<<<(BG * NN) / 256, 256, 0, stream>>>(dpart, ddeg);
  norm_kernel<<<(int)(((size_t)BG * NN * NN / 4) / 256), 256, 0, stream>>>(adj, ddeg, Abf);

  dim3 tb(32, 8);
  int cvt64 = BG * (NN / 32) * (FF / 32);
  int cvt32 = BG * (NN / 32) * (HH / 32);
  int ggrid = BG * (NN / 128);

  // ---- Layer 0 (f = 64) ----
  cvtT_kernel<<<cvt64, tb, 0, stream>>>(feats, XT, FF);
  gemm_kernel<4><<<ggrid, 128, 0, stream>>>(Abf, XT, Y1);
  cvtT_kernel<<<cvt64, tb, 0, stream>>>(Y1, XT, FF);
  gemm_kernel<4><<<ggrid, 128, 0, stream>>>(Abf, XT, Y2);
  cvtT_kernel<<<cvt64, tb, 0, stream>>>(Y2, XT, FF);
  gemm_kernel<4><<<ggrid, 128, 0, stream>>>(Abf, XT, Y3);
  proj_kernel<<<(BG * NN) / 8, dim3(HH, 8), 0, stream>>>(feats, Y1, Y2, Y3, W0, nullptr,
                                                         h0, FF, 4, HH, 0);

  // ---- Layer 1 (f = 32) ----
  cvtT_kernel<<<cvt32, tb, 0, stream>>>(h0, XT, HH);
  gemm_kernel<2><<<ggrid, 128, 0, stream>>>(Abf, XT, Y1);
  cvtT_kernel<<<cvt32, tb, 0, stream>>>(Y1, XT, HH);
  gemm_kernel<2><<<ggrid, 128, 0, stream>>>(Abf, XT, Y2);
  cvtT_kernel<<<cvt32, tb, 0, stream>>>(Y2, XT, HH);
  gemm_kernel<2><<<ggrid, 128, 0, stream>>>(Abf, XT, Y3);
  proj_kernel<<<(BG * NN) / 8, dim3(HH, 8), 0, stream>>>(h0, Y1, Y2, Y3, W1, nullptr,
                                                         h1, HH, 4, HH, 0);

  // ---- Layer 2 (f = 32) ----
  cvtT_kernel<<<cvt32, tb, 0, stream>>>(h1, XT, HH);
  gemm_kernel<2><<<ggrid, 128, 0, stream>>>(Abf, XT, Y1);
  cvtT_kernel<<<cvt32, tb, 0, stream>>>(Y1, XT, HH);
  gemm_kernel<2><<<ggrid, 128, 0, stream>>>(Abf, XT, Y2);
  cvtT_kernel<<<cvt32, tb, 0, stream>>>(Y2, XT, HH);
  gemm_kernel<2><<<ggrid, 128, 0, stream>>>(Abf, XT, Y3);
  proj_kernel<<<(BG * NN) / 8, dim3(HH, 8), 0, stream>>>(h1, Y1, Y2, Y3, W2, nullptr,
                                                         h2, HH, 4, HH, 0);

  // ---- Final dense: relu(concat(h0,h1,h2) @ Wm + bm) ----
  proj_kernel<<<(BG * NN) / 16, dim3(ODIM, 16), 0, stream>>>(h0, h1, h2, nullptr, Wm, bm,
                                                             out, HH, 3, ODIM, 1);
}
